// GraphormerModelV2_71159018160679
// MI455X (gfx1250) — compile-verified, matrix operand order checked by reading, not run on hardware
//
#include <hip/hip_runtime.h>
#include <math.h>

// Model dims (compile-time constants from the reference)
#define B_   16
#define N_   512
#define E_   8192
#define DIN_ 768
#define D_   256
#define H_   8
#define L_   4
#define DH_  32
#define FF_  1024

typedef _Float16 v16h __attribute__((ext_vector_type(16)));
typedef float    v8f  __attribute__((ext_vector_type(8)));
typedef unsigned int u32x4 __attribute__((ext_vector_type(4)));
typedef int          i32x4 __attribute__((ext_vector_type(4)));
typedef int          i32x8 __attribute__((ext_vector_type(8)));

#if __has_include(<hip/amd_detail/amd_gfx1250_TDM.h>)
#define TDM_6ARG 1
#endif

__device__ __forceinline__ float gelu_exact(float x) {
  return 0.5f * x * (1.0f + erff(x * 0.70710678118654752f));
}

// ---------------------------------------------------------------------------
// TDM: DMA a 2-D f32 tile [rows x cols] (row stride ld_elems) from global
// memory into LDS at byte offset lds_off. D# packing per ISA ch.8.
// ---------------------------------------------------------------------------
__device__ __forceinline__ void tdm_load_2d(const float* g, unsigned lds_off,
                                            int rows, int cols, int ld_elems) {
  unsigned long long ga = (unsigned long long)g;
  u32x4 g0;
  g0[0] = 1u;                                      // count=1, user descriptor
  g0[1] = lds_off;                                 // lds_addr (bytes)
  g0[2] = (unsigned)(ga & 0xFFFFFFFFu);            // global_addr[31:0]
  g0[3] = (unsigned)((ga >> 32) & 0x01FFFFFFu)     // global_addr[56:32]
        | 0x80000000u;                             // type=2 ("image")
  unsigned td0 = 1u << 30;                         // huge tensor dims: no clip
  unsigned td1 = 1u << 20;
  unsigned long long s0 = (unsigned long long)ld_elems;  // tensor_dim0_stride
  i32x8 g1;
  g1[0] = 0x00020000;                              // wg_mask=0, data_size=2 (4B)
  g1[1] = (int)((td0 & 0xFFFFu) << 16);            // tensor_dim0[15:0]
  g1[2] = (int)((td0 >> 16) | ((td1 & 0xFFFFu) << 16));
  g1[3] = (int)((td1 >> 16) | ((unsigned)(cols & 0xFFFF) << 16)); // tile_dim0
  g1[4] = (int)(rows & 0xFFFF);                    // tile_dim1 (tile_dim2=0)
  g1[5] = (int)(s0 & 0xFFFFFFFFu);                 // tensor_dim0_stride[31:0]
  g1[6] = (int)((s0 >> 32) & 0xFFFFu);             // stride[47:32], dim1_stride=0
  g1[7] = 0;
  i32x4 z4 = {0, 0, 0, 0};
#ifdef TDM_6ARG
  i32x8 z8 = {0, 0, 0, 0, 0, 0, 0, 0};
  __builtin_amdgcn_tensor_load_to_lds(g0, g1, z4, z4, z8, 0);
#else
  __builtin_amdgcn_tensor_load_to_lds(g0, g1, z4, z4, 0);
#endif
}

// Load one WMMA 16x16x32 f16 fragment from a row-major f32 matrix.
// A operand: lane m = row; B operand: lane n = row of W (i.e., column of B).
// Per-lane K ordering (ISA 7.12.2, 16-bit A 16x32):
//   lane<16 : elems 0..7 -> k0+0..7,  elems 8..15 -> k0+16..23
//   lane>=16: elems 0..7 -> k0+8..15, elems 8..15 -> k0+24..31
__device__ __forceinline__ v16h load_frag_f32(const float* __restrict__ base, int ld,
                                              int row0, int k0, int lane) {
  int m  = lane & 15;
  int kh = lane >> 4;
  const float* p = base + (size_t)(row0 + m) * ld + k0 + (kh << 3);
  v16h f;
#pragma unroll
  for (int e = 0; e < 8; ++e) f[e] = (_Float16)p[e];
#pragma unroll
  for (int e = 0; e < 8; ++e) f[8 + e] = (_Float16)p[16 + e];
  return f;
}

// ---------------------------------------------------------------------------
// C = X @ W^T + bias (optional exact GELU). X:[M,K] W:[N,K] C:[M,N].
// Block = 8 waves = 128x64 output tile; wave = 16 rows x 64 cols (4 WMMAs).
// W K-chunk tiles (64x32 f32) staged in LDS by the Tensor Data Mover.
// ---------------------------------------------------------------------------
#define BM_ 128
#define BN_ 64
__global__ void __launch_bounds__(256) gemm_wmma_kernel(
    const float* __restrict__ X, const float* __restrict__ W,
    const float* __restrict__ bias, float* __restrict__ C,
    int M, int K, int Nout, int act) {
  __shared__ float wt[BN_ * 32];  // 8 KB weight tile
  int lane = threadIdx.x;
  int wave = threadIdx.y;  // 0..7
  int nb = Nout / BN_;
  int bm = blockIdx.x / nb;
  int bn = blockIdx.x - bm * nb;
  int row0 = bm * BM_ + wave * 16;
  int m = lane & 15, kh = lane >> 4;
  unsigned lds_off = (unsigned)(unsigned long long)(void*)wt;

  v8f acc[4] = {};
  for (int k0 = 0; k0 < K; k0 += 32) {
    __syncthreads();  // previous chunk fully consumed
    if (wave == 0) {
      tdm_load_2d(W + (size_t)(bn * BN_) * K + k0, lds_off, BN_, 32, K);
      __builtin_amdgcn_s_wait_tensorcnt(0);
    }
    __syncthreads();  // tile visible to all waves
    v16h a = load_frag_f32(X, K, row0, k0, lane);
#pragma unroll
    for (int j = 0; j < 4; ++j) {
      const float* p = wt + (j * 16 + m) * 32 + (kh << 3);
      v16h bf;
#pragma unroll
      for (int e = 0; e < 8; ++e) bf[e] = (_Float16)p[e];
#pragma unroll
      for (int e = 0; e < 8; ++e) bf[8 + e] = (_Float16)p[16 + e];
      acc[j] = __builtin_amdgcn_wmma_f32_16x16x32_f16(false, a, false, bf,
                                                      (short)0, acc[j], false,
                                                      false);
    }
  }
  // C/D layout: VGPR r, lanes 0-15 -> row r, lanes 16-31 -> row r+8
  int rbase = row0 + (kh << 3);
#pragma unroll
  for (int j = 0; j < 4; ++j) {
    int col = bn * BN_ + j * 16 + m;
    float bv = bias ? bias[col] : 0.0f;
#pragma unroll
    for (int r = 0; r < 8; ++r) {
      float v = acc[j][r] + bv;
      if (act) v = gelu_exact(v);
      C[(size_t)(rbase + r) * Nout + col] = v;
    }
  }
}

// ---------------------------------------------------------------------------
// Attention: one wave per (b, h, 16-row q tile).
//   scores = (Q K^T)/sqrt(32) + bias, mask, softmax, out = P V (normalized).
// DH=32 => one v_wmma_f32_16x16x32_f16 per 16x16 score tile.
// ---------------------------------------------------------------------------
__global__ void __launch_bounds__(32) attn_kernel(
    const float* __restrict__ qkv, const float* __restrict__ bias,
    const unsigned char* __restrict__ nmask, float* __restrict__ outp) {
  __shared__ float s[16 * N_];
  __shared__ float invs[16];
  int lane = threadIdx.x;
  int qt = blockIdx.x & 31;
  int h  = (blockIdx.x >> 5) & 7;
  int b  = blockIdx.x >> 8;
  int m  = lane & 15;
  int kh = lane >> 4;

  const float* qbase = qkv + (size_t)b * N_ * (3 * D_) + h * DH_;
  const float* kbase = qbase + D_;
  const float* vbase = qbase + 2 * D_;
  const float* bb = bias + (((size_t)(b * H_ + h)) * N_ + (qt << 4)) * N_;
  const float scale = 0.17677669529663688f;  // 1/sqrt(32)

  v16h aq = load_frag_f32(qbase, 3 * D_, qt << 4, 0, lane);

  // ---- scores ----
  for (int kt = 0; kt < N_ / 16; ++kt) {
    v16h bk = load_frag_f32(kbase, 3 * D_, kt << 4, 0, lane);
    v8f c = {};
    c = __builtin_amdgcn_wmma_f32_16x16x32_f16(false, aq, false, bk, (short)0, c,
                                               false, false);
    int col = (kt << 4) + m;
    bool keep = nmask[b * N_ + col] != 0;
    int r0 = kh << 3;
#pragma unroll
    for (int r = 0; r < 8; ++r) {
      float v = c[r] * scale + bb[(size_t)(r0 + r) * N_ + col];
      if (!keep) v = -1e30f;
      s[(r0 + r) * N_ + col] = v;
    }
  }
  __syncthreads();

  // ---- softmax (2 lanes per row, 256 cols each) ----
  {
    float* sr = s + m * N_ + kh * 256;
    float mx = -3.4e38f;
    for (int i = 0; i < 256; ++i) mx = fmaxf(mx, sr[i]);
    mx = fmaxf(mx, __shfl_xor(mx, 16, 32));
    float sum = 0.0f;
    for (int i = 0; i < 256; ++i) {
      float e = __expf(sr[i] - mx);
      sr[i] = e;
      sum += e;
    }
    sum += __shfl_xor(sum, 16, 32);
    if (lane < 16) invs[lane] = 1.0f / sum;
  }
  __syncthreads();

  // ---- out = P V : 16x32 output, 2 WMMA accumulators, K=512 ----
  v8f c0 = {}, c1 = {};
  for (int k0 = 0; k0 < N_; k0 += 32) {
    v16h ap;
    {
      const float* p = s + m * N_ + k0 + (kh << 3);
#pragma unroll
      for (int e = 0; e < 8; ++e) ap[e] = (_Float16)p[e];
#pragma unroll
      for (int e = 0; e < 8; ++e) ap[8 + e] = (_Float16)p[16 + e];
    }
    v16h bv0, bv1;
#pragma unroll
    for (int e = 0; e < 16; ++e) {
      int kk = k0 + ((e < 8) ? e : e + 8) + (kh << 3);
      const float* vp = vbase + (size_t)kk * (3 * D_);
      bv0[e] = (_Float16)vp[m];
      bv1[e] = (_Float16)vp[16 + m];
    }
    c0 = __builtin_amdgcn_wmma_f32_16x16x32_f16(false, ap, false, bv0, (short)0,
                                                c0, false, false);
    c1 = __builtin_amdgcn_wmma_f32_16x16x32_f16(false, ap, false, bv1, (short)0,
                                                c1, false, false);
  }
  int r0 = kh << 3;
#pragma unroll
  for (int r = 0; r < 8; ++r) {
    int rr = r0 + r;
    float iv = invs[rr];
    size_t o = ((size_t)b * N_ + (qt << 4) + rr) * D_ + h * DH_;
    outp[o + m]      = c0[r] * iv;
    outp[o + 16 + m] = c1[r] * iv;
  }
}

// ---------------------------------------------------------------------------
// LayerNorm: out = LN(x + res_scale*y). One block per row of 256.
// ---------------------------------------------------------------------------
__global__ void __launch_bounds__(256) ln_kernel(
    const float* __restrict__ x, const float* __restrict__ y,
    const float* __restrict__ res_scale, const float* __restrict__ g,
    const float* __restrict__ bta, float* __restrict__ out) {
  int row = blockIdx.x, t = threadIdx.x;
  size_t o = (size_t)row * D_ + t;
  float v = x[o];
  if (y) v += res_scale[0] * y[o];
  __shared__ float red[D_];
  red[t] = v;
  __syncthreads();
  for (int s2 = 128; s2 > 0; s2 >>= 1) {
    if (t < s2) red[t] += red[t + s2];
    __syncthreads();
  }
  float mean = red[0] * (1.0f / D_);
  __syncthreads();
  float d = v - mean;
  red[t] = d * d;
  __syncthreads();
  for (int s2 = 128; s2 > 0; s2 >>= 1) {
    if (t < s2) red[t] += red[t + s2];
    __syncthreads();
  }
  float var = red[0] * (1.0f / D_);
  out[o] = d * rsqrtf(var + 1e-5f) * g[t] + bta[t];
}

__global__ void zero_f32(float* p, long long n) {
  long long i = (long long)blockIdx.x * blockDim.x + threadIdx.x;
  long long st = (long long)gridDim.x * blockDim.x;
  for (; i < n; i += st) p[i] = 0.0f;
}

__global__ void scatter_bias_kernel(const int* __restrict__ eidx,
                                    const int* __restrict__ etype,
                                    const float* __restrict__ ew,
                                    const float* __restrict__ emb,
                                    float* __restrict__ bias) {
  int i = blockIdx.x * blockDim.x + threadIdx.x;
  if (i >= B_ * E_) return;
  int b = i >> 13, e = i & (E_ - 1);
  int src = eidx[(b * 2 + 0) * E_ + e];
  int dst = eidx[(b * 2 + 1) * E_ + e];
  int t = etype[b * E_ + e];
  float extra = (t == 2) ? ew[b * E_ + e] : 0.0f;  // EDGE_TEXT_IMG
  float* bp = bias + (((size_t)b * H_) * N_ + src) * N_ + dst;
#pragma unroll
  for (int h = 0; h < H_; ++h)
    atomicAdd(bp + (size_t)h * N_ * N_, emb[t * H_ + h] + extra);
}

__global__ void pool_kernel(const float* __restrict__ xf,
                            const unsigned char* __restrict__ tm,
                            const unsigned char* __restrict__ im,
                            float* __restrict__ tp, float* __restrict__ ip) {
  int b = blockIdx.x, d = threadIdx.x;
  float st = 0, si = 0, ct = 0, ci = 0;
  for (int n = 0; n < N_; ++n) {
    float v = xf[((size_t)b * N_ + n) * D_ + d];
    float mt = tm[b * N_ + n] ? 1.0f : 0.0f;
    float mi = im[b * N_ + n] ? 1.0f : 0.0f;
    st += v * mt; ct += mt;
    si += v * mi; ci += mi;
  }
  tp[b * D_ + d] = st / fmaxf(ct, 1.0f);
  ip[b * D_ + d] = si / fmaxf(ci, 1.0f);
}

__global__ void gather_rows(const float* __restrict__ x, const int* __restrict__ gi,
                            float* __restrict__ dst) {
  int b = blockIdx.x, t = threadIdx.x;
  dst[b * D_ + t] = x[((size_t)b * N_ + gi[b]) * D_ + t];
}

__global__ void dense_kernel(const float* __restrict__ in, const float* __restrict__ W,
                             const float* __restrict__ bias, float* __restrict__ out,
                             int K, int O, int act) {
  int idx = blockIdx.x * blockDim.x + threadIdx.x;
  if (idx >= B_ * O) return;
  int b = idx / O, o = idx - b * O;
  const float* ip = in + (size_t)b * K;
  const float* wp = W + (size_t)o * K;
  float acc = bias ? bias[o] : 0.0f;
  for (int k = 0; k < K; ++k) acc = fmaf(ip[k], wp[k], acc);
  if (act) acc = gelu_exact(acc);
  out[idx] = acc;
}

__global__ void build_comb(const float* __restrict__ tf, const float* __restrict__ imf,
                           float* __restrict__ comb) {
  int b = blockIdx.x, t = threadIdx.x;
  float a = tf[b * D_ + t], c = imf[b * D_ + t];
  comb[b * 1024 + t]       = a;
  comb[b * 1024 + 256 + t] = a;
  comb[b * 1024 + 512 + t] = c;
  comb[b * 1024 + 768 + t] = c;
}

__global__ void build_combined(const float* __restrict__ lg, const float* __restrict__ tp,
                               const float* __restrict__ ip, const float* __restrict__ fg,
                               const float* __restrict__ conf, float* __restrict__ comb) {
  int b = blockIdx.x, t = threadIdx.x;
  const int KC = 7 * D_ + 1;  // 1793
#pragma unroll
  for (int l = 0; l < L_; ++l)
    comb[b * KC + l * D_ + t] = lg[l * (B_ * D_) + b * D_ + t];
  comb[b * KC + 4 * D_ + t] = tp[b * D_ + t];
  comb[b * KC + 5 * D_ + t] = ip[b * D_ + t];
  comb[b * KC + 6 * D_ + t] = fg[b * D_ + t];
  if (t == 0) comb[b * KC + 7 * D_] = conf[b];
}

__global__ void emit_out(const float* __restrict__ logits, const float* __restrict__ conf,
                         const float* __restrict__ tp, const float* __restrict__ ip,
                         float* __restrict__ out) {
  int i = blockIdx.x * blockDim.x + threadIdx.x;
  int total = 2 * B_ + 2 * B_ * D_;
  if (i >= total) return;
  if (i < B_) out[i] = logits[i];
  else if (i < 2 * B_) out[i] = conf[i - B_];
  else if (i < 2 * B_ + B_ * D_) out[i] = tp[i - 2 * B_];
  else out[i] = ip[i - 2 * B_ - B_ * D_];
}

// ---------------------------------------------------------------------------
extern "C" void kernel_launch(void* const* d_in, const int* in_sizes, int n_in,
                              void* d_out, int out_size, void* d_ws, size_t ws_size,
                              hipStream_t stream) {
  (void)in_sizes; (void)n_in; (void)out_size; (void)ws_size;
  // --- inputs, setup_inputs() insertion order, params flattened recursively ---
  const float* node_feats = (const float*)d_in[0];
  const float* edge_weight = (const float*)d_in[1];
  int pi = 2;
  const float* node_proj_W = (const float*)d_in[pi++];
  const float* node_proj_b = (const float*)d_in[pi++];
  const float* np_g = (const float*)d_in[pi++];
  const float* np_b = (const float*)d_in[pi++];
  const float* edge_type_emb = (const float*)d_in[pi++];
  struct Layer {
    const float *qkv_W, *qkv_b, *out_W, *out_b, *ln1_g, *ln1_b, *ln2_g, *ln2_b;
    const float *ff1_W, *ff1_b, *ff2_W, *ff2_b, *res_scale;
  } lyr[L_];
  for (int l = 0; l < L_; ++l) {
    lyr[l].qkv_W = (const float*)d_in[pi++]; lyr[l].qkv_b = (const float*)d_in[pi++];
    lyr[l].out_W = (const float*)d_in[pi++]; lyr[l].out_b = (const float*)d_in[pi++];
    lyr[l].ln1_g = (const float*)d_in[pi++]; lyr[l].ln1_b = (const float*)d_in[pi++];
    lyr[l].ln2_g = (const float*)d_in[pi++]; lyr[l].ln2_b = (const float*)d_in[pi++];
    lyr[l].ff1_W = (const float*)d_in[pi++]; lyr[l].ff1_b = (const float*)d_in[pi++];
    lyr[l].ff2_W = (const float*)d_in[pi++]; lyr[l].ff2_b = (const float*)d_in[pi++];
    lyr[l].res_scale = (const float*)d_in[pi++];
  }
  const float* fn_g = (const float*)d_in[pi++];
  const float* fn_b = (const float*)d_in[pi++];
  const float* cd_text_W = (const float*)d_in[pi++];
  const float* cd_text_b = (const float*)d_in[pi++];
  const float* cd_img_W = (const float*)d_in[pi++];
  const float* cd_img_b = (const float*)d_in[pi++];
  const float* cd1_W = (const float*)d_in[pi++];
  const float* cd1_b = (const float*)d_in[pi++];
  const float* cd2_W = (const float*)d_in[pi++];
  const float* cd2_b = (const float*)d_in[pi++];
  const float* cd3_W = (const float*)d_in[pi++];
  const float* cd3_b = (const float*)d_in[pi++];
  const float* cls1_W = (const float*)d_in[pi++];
  const float* cls1_b = (const float*)d_in[pi++];
  const float* cls2_W = (const float*)d_in[pi++];
  const float* cls2_b = (const float*)d_in[pi++];
  const float* cls3_W = (const float*)d_in[pi++];
  const float* cls3_b = (const float*)d_in[pi++];
  const unsigned char* node_mask  = (const unsigned char*)d_in[pi++];
  const unsigned char* text_mask  = (const unsigned char*)d_in[pi++];
  const unsigned char* image_mask = (const unsigned char*)d_in[pi++];
  const int* global_indices = (const int*)d_in[pi++];
  const int* edge_index = (const int*)d_in[pi++];
  const int* edge_type  = (const int*)d_in[pi++];

  // --- workspace layout (floats) ---
  float* ws = (float*)d_ws;
  const long long BIAS_SZ = (long long)B_ * H_ * N_ * N_;  // 33,554,432
  float* BIAS = ws;                 ws += BIAS_SZ;
  float* X    = ws;                 ws += (long long)B_ * N_ * D_;
  float* T0   = ws;                 ws += (long long)B_ * N_ * D_;
  float* QKV  = ws;                 ws += (long long)B_ * N_ * 3 * D_;
  float* FFH  = ws;                 ws += (long long)B_ * N_ * FF_;
  float* ATT  = ws;                 ws += (long long)B_ * N_ * D_;
  float* XF   = ws;                 ws += (long long)B_ * N_ * D_;
  float* LG   = ws;                 ws += (long long)L_ * B_ * D_;
  float* TP   = ws;                 ws += B_ * D_;
  float* IP   = ws;                 ws += B_ * D_;
  float* FG   = ws;                 ws += B_ * D_;
  float* TF   = ws;                 ws += B_ * D_;
  float* IMF  = ws;                 ws += B_ * D_;
  float* COMB = ws;                 ws += B_ * 4 * D_;
  float* HC1  = ws;                 ws += B_ * D_;
  float* HC2  = ws;                 ws += B_ * (D_ / 2);
  float* CONF = ws;                 ws += B_;
  float* CMB2 = ws;                 ws += B_ * (7 * D_ + 1);
  float* CH1  = ws;                 ws += B_ * 2 * D_;
  float* CH2  = ws;                 ws += B_ * D_;
  float* LOG  = ws;                 ws += B_;

  const int M = B_ * N_;  // 8192 rows

  auto gemm = [&](const float* Xm, const float* Wm, const float* bm, float* Cm,
                  int K, int Nout, int act) {
    int blocks = (M / BM_) * (Nout / BN_);  // exact for all shapes used
    dim3 blk(32, 8);
    gemm_wmma_kernel<<<blocks, blk, 0, stream>>>(Xm, Wm, bm, Cm, M, K, Nout, act);
  };

  // 1) attention bias scatter
  zero_f32<<<2048, 256, 0, stream>>>(BIAS, BIAS_SZ);
  scatter_bias_kernel<<<(B_ * E_ + 255) / 256, 256, 0, stream>>>(
      edge_index, edge_type, edge_weight, edge_type_emb, BIAS);

  // 2) node projection + LN
  gemm(node_feats, node_proj_W, node_proj_b, T0, DIN_, D_, 0);
  ln_kernel<<<M, 256, 0, stream>>>(T0, nullptr, nullptr, np_g, np_b, X);

  // 3) transformer layers
  for (int l = 0; l < L_; ++l) {
    gemm(X, lyr[l].qkv_W, lyr[l].qkv_b, QKV, D_, 3 * D_, 0);
    attn_kernel<<<B_ * H_ * (N_ / 16), 32, 0, stream>>>(QKV, BIAS, node_mask, ATT);
    gemm(ATT, lyr[l].out_W, lyr[l].out_b, T0, D_, D_, 0);
    ln_kernel<<<M, 256, 0, stream>>>(X, T0, lyr[l].res_scale, lyr[l].ln1_g,
                                     lyr[l].ln1_b, X);
    gemm(X, lyr[l].ff1_W, lyr[l].ff1_b, FFH, D_, FF_, 1);
    gemm(FFH, lyr[l].ff2_W, lyr[l].ff2_b, T0, FF_, D_, 0);
    ln_kernel<<<M, 256, 0, stream>>>(X, T0, lyr[l].res_scale, lyr[l].ln2_g,
                                     lyr[l].ln2_b, X);
    gather_rows<<<B_, 256, 0, stream>>>(X, global_indices, LG + l * (B_ * D_));
  }

  // 4) final LN, pools, global gather
  ln_kernel<<<M, 256, 0, stream>>>(X, nullptr, nullptr, fn_g, fn_b, XF);
  pool_kernel<<<B_, 256, 0, stream>>>(XF, text_mask, image_mask, TP, IP);
  gather_rows<<<B_, 256, 0, stream>>>(XF, global_indices, FG);

  // 5) conflict head
  auto dense = [&](const float* in, const float* Wm, const float* bm, float* out,
                   int K, int O, int act) {
    dense_kernel<<<(B_ * O + 255) / 256, 256, 0, stream>>>(in, Wm, bm, out, K, O,
                                                           act);
  };
  dense(TP, cd_text_W, cd_text_b, TF, D_, D_, 0);
  dense(IP, cd_img_W, cd_img_b, IMF, D_, D_, 0);
  build_comb<<<B_, 256, 0, stream>>>(TF, IMF, COMB);
  dense(COMB, cd1_W, cd1_b, HC1, 4 * D_, D_, 1);
  dense(HC1, cd2_W, cd2_b, HC2, D_, D_ / 2, 1);
  dense(HC2, cd3_W, cd3_b, CONF, D_ / 2, 1, 0);

  // 6) classifier head
  build_combined<<<B_, 256, 0, stream>>>(LG, TP, IP, FG, CONF, CMB2);
  dense(CMB2, cls1_W, cls1_b, CH1, 7 * D_ + 1, 2 * D_, 1);
  dense(CH1, cls2_W, cls2_b, CH2, 2 * D_, D_, 1);
  dense(CH2, cls3_W, cls3_b, LOG, D_, 1, 0);

  // 7) outputs: [logits(16) | conflict(16) | text_pool(4096) | image_pool(4096)]
  emit_out<<<(2 * B_ + 2 * B_ * D_ + 255) / 256, 256, 0, stream>>>(
      LOG, CONF, TP, IP, (float*)d_out);
}
